// MultipleHeadAttention_84731114816253
// MI455X (gfx1250) — matrix operand order repaired and checked by reading, not compiled, not run on hardware
//
#include <hip/hip_runtime.h>
#include <math.h>

// ---------------------------------------------------------------------------
// Types for CDNA5 WMMA (wave32): 16x16x32 bf16 -> f32 accumulate
// ---------------------------------------------------------------------------
typedef __attribute__((ext_vector_type(16))) __bf16 v16bf;
typedef __attribute__((ext_vector_type(8)))  __bf16 bf16x8;
typedef __attribute__((ext_vector_type(8)))  float  v8f;
typedef __attribute__((ext_vector_type(4)))  int    i32x4;

#define B_   8
#define S_   2048
#define D_   512
#define H_   8
#define DH_  64
#define HID_ 512
#define FF_  1024
#define M_TOTAL (B_ * S_)   // 16384

// gfx1250 async global->LDS copy path (ASYNCcnt), guarded for compile safety.
#if defined(__gfx1250__) && __has_builtin(__builtin_amdgcn_global_load_async_to_lds_b128) && \
    __has_builtin(__builtin_amdgcn_s_wait_asynccnt)
#define USE_ASYNC_COPY 1
#else
#define USE_ASYNC_COPY 0
#endif

static __device__ __forceinline__ void async_copy_b128(void* lds_dst, const void* gsrc) {
#if USE_ASYNC_COPY
  __builtin_amdgcn_global_load_async_to_lds_b128(
      (__attribute__((address_space(1))) i32x4*)(gsrc),
      (__attribute__((address_space(3))) i32x4*)(lds_dst), 0, 0);
#else
  (void)lds_dst; (void)gsrc;
#endif
}
static __device__ __forceinline__ void wait_async_all() {
#if USE_ASYNC_COPY
  __builtin_amdgcn_s_wait_asynccnt(0);
#endif
}

static __device__ __forceinline__ v8f vzero8() {
  v8f z;
#pragma unroll
  for (int i = 0; i < 8; ++i) z[i] = 0.0f;
  return z;
}

static __device__ __forceinline__ v8f wmma_bf16(v16bf a, v16bf b, v8f c) {
  return __builtin_amdgcn_wmma_f32_16x16x32_bf16(false, a, false, b, (short)0, c,
                                                 false, false);
}

// A fragment, 16x32 bf16, from row-major [16][K] source.
// lane L: m=L&15; koff=(L>>4)*8; elems 0..7 -> K=koff+j, 8..15 -> K=koff+16+(j-8)
static __device__ __forceinline__ v16bf load_frag_a(const __bf16* base, int stride,
                                                    int lane) {
  const __bf16* p = base + (size_t)(lane & 15) * stride + ((lane >> 4) << 3);
  bf16x8 c0 = *(const bf16x8*)(p);
  bf16x8 c1 = *(const bf16x8*)(p + 16);
  v16bf r;
#pragma unroll
  for (int j = 0; j < 8; ++j) { r[j] = c0[j]; r[j + 8] = c1[j]; }
  return r;
}

// B fragment, 32x16 bf16 (K x N), from [N][K] laid source (K contiguous).
// lane L: n=L&15; koff=(L>>4)*8; elems 0..7 -> K=koff+j, 8..15 -> K=koff+16+(j-8)
static __device__ __forceinline__ v16bf load_frag_b(const __bf16* base, int stride,
                                                    int lane) {
  const __bf16* p = base + (size_t)(lane & 15) * stride + ((lane >> 4) << 3);
  bf16x8 c0 = *(const bf16x8*)(p);
  bf16x8 c1 = *(const bf16x8*)(p + 16);
  v16bf r;
#pragma unroll
  for (int j = 0; j < 8; ++j) { r[j] = c0[j]; r[j + 8] = c1[j]; }
  return r;
}

// XOR-butterfly broadcast within a 16-lane group via ds_swizzle (imm pattern).
template <int M>
static __device__ __forceinline__ float swz_xor(float v) {
#if defined(__gfx1250__) && __has_builtin(__builtin_amdgcn_ds_swizzle)
  return __int_as_float(
      __builtin_amdgcn_ds_swizzle(__float_as_int(v), (M << 10) | 0x1f));
#else
  return __shfl_xor(v, M, 32);
#endif
}
static __device__ __forceinline__ void red_max8(float* a) {
#pragma unroll
  for (int r = 0; r < 8; ++r) {
    float x = a[r];
    x = fmaxf(x, swz_xor<1>(x));
    x = fmaxf(x, swz_xor<2>(x));
    x = fmaxf(x, swz_xor<4>(x));
    x = fmaxf(x, swz_xor<8>(x));
    a[r] = x;
  }
}
static __device__ __forceinline__ void red_sum8(float* a) {
#pragma unroll
  for (int r = 0; r < 8; ++r) {
    float x = a[r];
    x += swz_xor<1>(x);
    x += swz_xor<2>(x);
    x += swz_xor<4>(x);
    x += swz_xor<8>(x);
    a[r] = x;
  }
}

// ---------------------------------------------------------------------------
// Elementwise casts / weight prep
// ---------------------------------------------------------------------------
__global__ void cast_to_bf16_k(const float* __restrict__ in, __bf16* __restrict__ out,
                               int n) {
  int i = blockIdx.x * blockDim.x + threadIdx.x;
  if (i < n) out[i] = (__bf16)in[i];
}

__global__ void build_qkv_k(const float* __restrict__ qw, const float* __restrict__ kw,
                            const float* __restrict__ vw, const float* __restrict__ qb,
                            const float* __restrict__ kb, const float* __restrict__ vb,
                            __bf16* __restrict__ wout, float* __restrict__ bout) {
  int i = blockIdx.x * blockDim.x + threadIdx.x;
  const int WSZ = HID_ * D_;  // 262144
  if (i < 3 * WSZ) {
    float v = (i < WSZ) ? qw[i] : (i < 2 * WSZ ? kw[i - WSZ] : vw[i - 2 * WSZ]);
    wout[i] = (__bf16)v;
  }
  if (i < 3 * HID_) {
    bout[i] = (i < HID_) ? qb[i] : (i < 2 * HID_ ? kb[i - HID_] : vb[i - 2 * HID_]);
  }
}

// ---------------------------------------------------------------------------
// WMMA GEMM: C[M,N] = act(A[M,K] @ W[N,K]^T + bias) (+res)
// Block tile 128x128, 8 waves (4x2), 32x64 per wave, K step 32.
// Double-buffered LDS; async global->LDS copies when available.
// ---------------------------------------------------------------------------
#define EPI_SILU 1
#define EPI_RES  2
#define EPI_F32  4

template <int EPI>
__global__ __launch_bounds__(256) void gemm_bf16_wmma(
    const __bf16* __restrict__ A, const __bf16* __restrict__ W,
    const float* __restrict__ bias, const float* __restrict__ res,
    __bf16* __restrict__ outb, float* __restrict__ outf, int M, int N, int K) {
  constexpr int BM = 128, BN = 128, BK = 32;
  constexpr int LDT = BK + 8;  // 40
  __shared__ __attribute__((aligned(16))) __bf16 at[2][BM * LDT];
  __shared__ __attribute__((aligned(16))) __bf16 bt[2][BN * LDT];

  const int tid = threadIdx.x;
  const int lane = tid & 31;
  const int wave = tid >> 5;
  const int wm = (wave & 3) * 32;        // 4 waves along M
  const int wn = (wave >> 2) * 64;       // 2 waves along N
  const int m0 = blockIdx.y * BM;
  const int n0 = blockIdx.x * BN;

  // per-thread tile-copy coordinates: 16 bf16 per thread per tile
  const int lr = tid >> 1;               // 0..127
  const int lc = (tid & 1) * 16;         // 0 / 16
  const __bf16* ga = A + (size_t)(m0 + lr) * K + lc;
  const __bf16* gw = W + (size_t)(n0 + lr) * K + lc;
  const int lds_off = lr * LDT + lc;

  v8f c[2][4];
#pragma unroll
  for (int i = 0; i < 2; ++i)
#pragma unroll
    for (int j = 0; j < 4; ++j) c[i][j] = vzero8();

#if USE_ASYNC_COPY
  // ---- async double-buffered pipeline ----
  async_copy_b128(&at[0][lds_off], ga);
  async_copy_b128(&at[0][lds_off + 8], ga + 8);
  async_copy_b128(&bt[0][lds_off], gw);
  async_copy_b128(&bt[0][lds_off + 8], gw + 8);
  wait_async_all();
  __syncthreads();
  int cur = 0;
  for (int k0 = 0; k0 < K; k0 += BK) {
    if (k0 + BK < K) {  // issue next tile into the other buffer
      async_copy_b128(&at[cur ^ 1][lds_off], ga + k0 + BK);
      async_copy_b128(&at[cur ^ 1][lds_off + 8], ga + k0 + BK + 8);
      async_copy_b128(&bt[cur ^ 1][lds_off], gw + k0 + BK);
      async_copy_b128(&bt[cur ^ 1][lds_off + 8], gw + k0 + BK + 8);
    }
    if (k0 + 2 * BK < K) {
      __builtin_prefetch(ga + k0 + 2 * BK, 0, 1);
      __builtin_prefetch(gw + k0 + 2 * BK, 0, 1);
    }
    v16bf a0 = load_frag_a(&at[cur][(wm + 0) * LDT], LDT, lane);
    v16bf a1 = load_frag_a(&at[cur][(wm + 16) * LDT], LDT, lane);
#pragma unroll
    for (int j = 0; j < 4; ++j) {
      v16bf bf = load_frag_b(&bt[cur][(wn + j * 16) * LDT], LDT, lane);
      c[0][j] = wmma_bf16(a0, bf, c[0][j]);
      c[1][j] = wmma_bf16(a1, bf, c[1][j]);
    }
    wait_async_all();
    __syncthreads();
    cur ^= 1;
  }
#else
  // ---- synchronous fallback: register-staged double buffering ----
  bf16x8 ra0 = *(const bf16x8*)(ga);
  bf16x8 ra1 = *(const bf16x8*)(ga + 8);
  bf16x8 rw0 = *(const bf16x8*)(gw);
  bf16x8 rw1 = *(const bf16x8*)(gw + 8);
  *(bf16x8*)&at[0][lds_off] = ra0;
  *(bf16x8*)&at[0][lds_off + 8] = ra1;
  *(bf16x8*)&bt[0][lds_off] = rw0;
  *(bf16x8*)&bt[0][lds_off + 8] = rw1;
  __syncthreads();
  int cur = 0;
  for (int k0 = 0; k0 < K; k0 += BK) {
    const bool more = (k0 + BK < K);
    if (more) {
      ra0 = *(const bf16x8*)(ga + k0 + BK);
      ra1 = *(const bf16x8*)(ga + k0 + BK + 8);
      rw0 = *(const bf16x8*)(gw + k0 + BK);
      rw1 = *(const bf16x8*)(gw + k0 + BK + 8);
      if (k0 + 2 * BK < K) {
        __builtin_prefetch(ga + k0 + 2 * BK, 0, 1);
        __builtin_prefetch(gw + k0 + 2 * BK, 0, 1);
      }
    }
    v16bf a0 = load_frag_a(&at[cur][(wm + 0) * LDT], LDT, lane);
    v16bf a1 = load_frag_a(&at[cur][(wm + 16) * LDT], LDT, lane);
#pragma unroll
    for (int j = 0; j < 4; ++j) {
      v16bf bf = load_frag_b(&bt[cur][(wn + j * 16) * LDT], LDT, lane);
      c[0][j] = wmma_bf16(a0, bf, c[0][j]);
      c[1][j] = wmma_bf16(a1, bf, c[1][j]);
    }
    if (more) {
      *(bf16x8*)&at[cur ^ 1][lds_off] = ra0;
      *(bf16x8*)&at[cur ^ 1][lds_off + 8] = ra1;
      *(bf16x8*)&bt[cur ^ 1][lds_off] = rw0;
      *(bf16x8*)&bt[cur ^ 1][lds_off + 8] = rw1;
    }
    __syncthreads();
    cur ^= 1;
  }
#endif

  // C/D layout: elem r -> M = fragM + r + 8*(lane>>4), N = fragN + (lane&15)
#pragma unroll
  for (int i = 0; i < 2; ++i)
#pragma unroll
    for (int j = 0; j < 4; ++j) {
      int n = n0 + wn + j * 16 + (lane & 15);
      float bv = bias[n];
#pragma unroll
      for (int r = 0; r < 8; ++r) {
        int m = m0 + wm + i * 16 + r + ((lane >> 4) << 3);
        float v = c[i][j][r] + bv;
        if constexpr (EPI & EPI_SILU) v = v / (1.0f + __expf(-v));
        if constexpr (EPI & EPI_RES) v += res[(size_t)m * N + n];
        if constexpr (EPI & EPI_F32)
          outf[(size_t)m * N + n] = v;
        else
          outb[(size_t)m * N + n] = (__bf16)v;
      }
    }
}

// ---------------------------------------------------------------------------
// Flash attention: one workgroup (4 waves) per (b, h, 64-row q-tile).
// Each wave owns 16 q rows; KV streamed 32 rows/iter through LDS.
// ---------------------------------------------------------------------------
__global__ __launch_bounds__(128) void attention_wmma_k(
    const __bf16* __restrict__ qkv,  // [B*S, 1536] = q | k | v
    float* __restrict__ o)           // [B*S, 512]
{
  constexpr int LDK = DH_ + 8;  // 72
  constexpr int LDV = 32 + 8;   // 40
  __shared__ __attribute__((aligned(16))) __bf16 kt[32 * LDK];   // [kv][dh]
  __shared__ __attribute__((aligned(16))) __bf16 vt[DH_ * LDV];  // [dh][kv]
  __shared__ __attribute__((aligned(16))) __bf16 pt[4][16 * 32]; // per-wave P

  const int tid = threadIdx.x;
  const int lane = tid & 31;
  const int wave = tid >> 5;
  const int qt = blockIdx.x;
  const int h = blockIdx.y;
  const int b = blockIdx.z;

  const int row0 = qt * 64 + wave * 16;
  const size_t qbase = ((size_t)b * S_ + row0) * 1536 + (size_t)h * DH_;

  v16bf qa0 = load_frag_a(qkv + qbase, 1536, lane);       // dh 0..31
  v16bf qa1 = load_frag_a(qkv + qbase + 32, 1536, lane);  // dh 32..63

  v8f acc[4];
#pragma unroll
  for (int f = 0; f < 4; ++f) acc[f] = vzero8();
  float rm[8], rl[8];
#pragma unroll
  for (int r = 0; r < 8; ++r) { rm[r] = -INFINITY; rl[r] = 0.0f; }

  const float scale = 0.125f;            // 1/sqrt(64)
  const int prow_lo = (lane >> 4) << 3;  // rows this lane holds in C frags

  // per-thread KV copy coordinates
  const int kr = tid >> 2;           // kv row 0..31
  const int kc = (tid & 3) * 16;     // dh chunk

  for (int kv0 = 0; kv0 < S_; kv0 += 32) {
    {  // cooperative tile load: K row-major (async if available), V transposed
      size_t rbase = ((size_t)b * S_ + kv0 + kr) * 1536 + (size_t)h * DH_;
      const __bf16* gk = qkv + rbase + 512 + kc;
      async_copy_b128(&kt[kr * LDK + kc], gk);
      async_copy_b128(&kt[kr * LDK + kc + 8], gk + 8);
#if !USE_ASYNC_COPY
      bf16x8 u0 = *(const bf16x8*)gk;
      bf16x8 u1 = *(const bf16x8*)(gk + 8);
      *(bf16x8*)&kt[kr * LDK + kc] = u0;
      *(bf16x8*)&kt[kr * LDK + kc + 8] = u1;
#endif
      const __bf16* gv = qkv + rbase + 1024 + kc;
      bf16x8 w0 = *(const bf16x8*)gv;
      bf16x8 w1 = *(const bf16x8*)(gv + 8);
#pragma unroll
      for (int j = 0; j < 8; ++j) {
        vt[(kc + j) * LDV + kr] = w0[j];
        vt[(kc + 8 + j) * LDV + kr] = w1[j];
      }
      wait_async_all();
    }
    __syncthreads();

    // scores = q @ k^T : two 16(q)x16(kv) tiles, K-dim = dh (2 x 32)
    v8f s0 = vzero8(), s1 = vzero8();
    s0 = wmma_bf16(qa0, load_frag_b(&kt[0 * LDK], LDK, lane), s0);
    s0 = wmma_bf16(qa1, load_frag_b(&kt[0 * LDK] + 32, LDK, lane), s0);
    s1 = wmma_bf16(qa0, load_frag_b(&kt[16 * LDK], LDK, lane), s1);
    s1 = wmma_bf16(qa1, load_frag_b(&kt[16 * LDK] + 32, LDK, lane), s1);

    // online softmax over this 32-wide KV slab
    float mloc[8];
#pragma unroll
    for (int r = 0; r < 8; ++r) {
      s0[r] *= scale;
      s1[r] *= scale;
      mloc[r] = fmaxf(s0[r], s1[r]);
    }
    red_max8(mloc);

    float alpha[8], ps[8];
#pragma unroll
    for (int r = 0; r < 8; ++r) {
      float mn = fmaxf(rm[r], mloc[r]);
      alpha[r] = __expf(rm[r] - mn);
      rm[r] = mn;
      float p0 = __expf(s0[r] - mn);
      float p1 = __expf(s1[r] - mn);
      ps[r] = p0 + p1;
      int pr = prow_lo + r;
      pt[wave][pr * 32 + (lane & 15)] = (__bf16)p0;
      pt[wave][pr * 32 + 16 + (lane & 15)] = (__bf16)p1;
    }
    red_sum8(ps);
#pragma unroll
    for (int r = 0; r < 8; ++r) rl[r] = rl[r] * alpha[r] + ps[r];
#pragma unroll
    for (int f = 0; f < 4; ++f)
#pragma unroll
      for (int r = 0; r < 8; ++r) acc[f][r] *= alpha[r];

    // pt is wave-private; same-wave LDS ops are in-order -> wave barrier only
    __builtin_amdgcn_wave_barrier();

    // O += P @ V : K-dim = kv (32), N = dh (4 x 16)
    v16bf pa = load_frag_a(pt[wave], 32, lane);
#pragma unroll
    for (int f = 0; f < 4; ++f) {
      v16bf vb = load_frag_b(&vt[(f * 16) * LDV], LDV, lane);
      acc[f] = wmma_bf16(pa, vb, acc[f]);
    }
    __syncthreads();  // protect kt/vt for next slab
  }

#pragma unroll
  for (int r = 0; r < 8; ++r) rl[r] = 1.0f / rl[r];
#pragma unroll
  for (int f = 0; f < 4; ++f) {
    int dh = f * 16 + (lane & 15);
#pragma unroll
    for (int r = 0; r < 8; ++r) {
      int srow = row0 + prow_lo + r;
      o[((size_t)b * S_ + srow) * HID_ + h * DH_ + dh] = acc[f][r] * rl[r];
    }
  }
}

// ---------------------------------------------------------------------------
// LayerNorm over 512 cols; writes f32 (residual) + bf16 (GEMM input) copies
// ---------------------------------------------------------------------------
__global__ __launch_bounds__(256) void layernorm_k(const float* __restrict__ x,
                                                   const float* __restrict__ g,
                                                   const float* __restrict__ bta,
                                                   float* __restrict__ yf,
                                                   __bf16* __restrict__ yb) {
  __shared__ float red[16];
  const int row = blockIdx.x;
  const int tid = threadIdx.x;
  const float* xr = x + (size_t)row * HID_;
  float v0 = xr[tid], v1 = xr[tid + 256];
  float s = v0 + v1, sq = v0 * v0 + v1 * v1;
#pragma unroll
  for (int m = 1; m < 32; m <<= 1) {
    s += __shfl_xor(s, m, 32);
    sq += __shfl_xor(sq, m, 32);
  }
  const int wave = tid >> 5, lane = tid & 31;
  if (lane == 0) { red[wave] = s; red[8 + wave] = sq; }
  __syncthreads();
  if (wave == 0) {
    float a = (lane < 8) ? red[lane] : 0.0f;
    float bq = (lane < 8) ? red[8 + lane] : 0.0f;
#pragma unroll
    for (int m = 1; m < 8; m <<= 1) {
      a += __shfl_xor(a, m, 32);
      bq += __shfl_xor(bq, m, 32);
    }
    if (lane == 0) { red[0] = a; red[8] = bq; }
  }
  __syncthreads();
  float mu = red[0] * (1.0f / HID_);
  float var = red[8] * (1.0f / HID_) - mu * mu;
  float rstd = rsqrtf(var + 1e-5f);
  float y0 = (v0 - mu) * rstd * g[tid] + bta[tid];
  float y1 = (v1 - mu) * rstd * g[tid + 256] + bta[tid + 256];
  size_t base = (size_t)row * HID_;
  yf[base + tid] = y0;
  yf[base + tid + 256] = y1;
  yb[base + tid] = (__bf16)y0;
  yb[base + tid + 256] = (__bf16)y1;
}

// ---------------------------------------------------------------------------
// Launch
// ---------------------------------------------------------------------------
extern "C" void kernel_launch(void* const* d_in, const int* in_sizes, int n_in,
                              void* d_out, int out_size, void* d_ws, size_t ws_size,
                              hipStream_t stream) {
  (void)in_sizes; (void)n_in; (void)out_size; (void)ws_size;
  const float* x    = (const float*)d_in[0];
  const float* qw   = (const float*)d_in[1];
  const float* qb   = (const float*)d_in[2];
  const float* kw   = (const float*)d_in[3];
  const float* kb   = (const float*)d_in[4];
  const float* vw   = (const float*)d_in[5];
  const float* vb   = (const float*)d_in[6];
  const float* ln_g = (const float*)d_in[7];
  const float* ln_b = (const float*)d_in[8];
  const float* w1   = (const float*)d_in[9];
  const float* b1   = (const float*)d_in[10];
  const float* w2   = (const float*)d_in[11];
  const float* b2   = (const float*)d_in[12];
  const float* fw   = (const float*)d_in[13];
  const float* fb   = (const float*)d_in[14];
  float* out = (float*)d_out;

  char* ws = (char*)d_ws;
  size_t off = 0;
  auto alloc = [&](size_t bytes) {
    void* p = ws + off;
    off = (off + bytes + 255) & ~(size_t)255;
    return p;
  };
  __bf16* x_b    = (__bf16*)alloc((size_t)M_TOTAL * D_ * 2);
  __bf16* wqkv_b = (__bf16*)alloc((size_t)1536 * D_ * 2);
  float*  bqkv   = (float*)alloc(1536 * 4);
  __bf16* w1_b   = (__bf16*)alloc((size_t)FF_ * HID_ * 2);
  __bf16* w2_b   = (__bf16*)alloc((size_t)HID_ * FF_ * 2);
  __bf16* fw_b   = (__bf16*)alloc((size_t)D_ * HID_ * 2);
  __bf16* qkv_b  = (__bf16*)alloc((size_t)M_TOTAL * 1536 * 2);
  float*  o_f    = (float*)alloc((size_t)M_TOTAL * HID_ * 4);
  float*  oln_f  = (float*)alloc((size_t)M_TOTAL * HID_ * 4);
  __bf16* oln_b  = (__bf16*)alloc((size_t)M_TOTAL * HID_ * 2);
  __bf16* h1_b   = (__bf16*)alloc((size_t)M_TOTAL * FF_ * 2);
  __bf16* h_b    = (__bf16*)alloc((size_t)M_TOTAL * HID_ * 2);

  cast_to_bf16_k<<<(M_TOTAL * D_ + 255) / 256, 256, 0, stream>>>(x, x_b, M_TOTAL * D_);
  build_qkv_k<<<(3 * HID_ * D_ + 255) / 256, 256, 0, stream>>>(qw, kw, vw, qb, kb, vb,
                                                               wqkv_b, bqkv);
  cast_to_bf16_k<<<(FF_ * HID_ + 255) / 256, 256, 0, stream>>>(w1, w1_b, FF_ * HID_);
  cast_to_bf16_k<<<(HID_ * FF_ + 255) / 256, 256, 0, stream>>>(w2, w2_b, HID_ * FF_);
  cast_to_bf16_k<<<(D_ * HID_ + 255) / 256, 256, 0, stream>>>(fw, fw_b, D_ * HID_);

  // Fused QKV projection: [16384,512] x [1536,512]^T
  gemm_bf16_wmma<0><<<dim3(1536 / 128, M_TOTAL / 128), 256, 0, stream>>>(
      x_b, wqkv_b, bqkv, nullptr, qkv_b, nullptr, M_TOTAL, 1536, 512);

  attention_wmma_k<<<dim3(S_ / 64, H_, B_), 128, 0, stream>>>(qkv_b, o_f);

  layernorm_k<<<M_TOTAL, 256, 0, stream>>>(o_f, ln_g, ln_b, oln_f, oln_b);

  // FFN: silu(o @ w1^T + b1) @ w2^T + b2 + o
  gemm_bf16_wmma<EPI_SILU><<<dim3(FF_ / 128, M_TOTAL / 128), 256, 0, stream>>>(
      oln_b, w1_b, b1, nullptr, h1_b, nullptr, M_TOTAL, FF_, HID_);
  gemm_bf16_wmma<EPI_RES><<<dim3(HID_ / 128, M_TOTAL / 128), 256, 0, stream>>>(
      h1_b, w2_b, b2, oln_f, h_b, nullptr, M_TOTAL, HID_, FF_);

  // out = x + h @ fw^T + fb  (f32 output)
  gemm_bf16_wmma<EPI_RES | EPI_F32><<<dim3(D_ / 128, M_TOTAL / 128), 256, 0, stream>>>(
      h_b, fw_b, fb, x, nullptr, out, M_TOTAL, D_, HID_);
}
